// GATRelateCNet_22101901705654
// MI455X (gfx1250) — matrix operand order repaired and checked by reading, not compiled, not run on hardware
//
#include <hip/hip_runtime.h>
#include <hip/hip_bf16.h>
#include <math.h>

// ---------------------------------------------------------------------------
// GATRelateCNet forward for MI455X (gfx1250).
// Conv / transposed-conv layers = implicit GEMM on v_wmma_f32_16x16x32_f16
// (f16 A/B, f32 accumulate). Weight tiles are DMA'd to LDS with
// global_load_async_to_lds_b32 (ASYNCcnt), activation tiles staged in
// fragment-major order so fragments load as 2x ds_load_b128 per matrix.
// ---------------------------------------------------------------------------

#define B_    2
#define NN    300
#define NHALF 150
#define S0_   128
#define E_TOT 9600
#define DEG_  16
#define BN_SC 0.9999950000374997f   // 1/sqrt(1+1e-5)

typedef __attribute__((ext_vector_type(16))) _Float16 v16h;
typedef __attribute__((ext_vector_type(8)))  float    v8f;

__device__ __forceinline__ float eluf(float x) { return x > 0.f ? x : (expf(x) - 1.f); }

// LDS byte offset of a pointer into __shared__ (AS3 pointers are the
// wave-relative LDS byte address the async-to-LDS VDST operand expects).
__device__ __forceinline__ unsigned lds_off_u32(const void* p) {
  return (unsigned)(uintptr_t)(__attribute__((address_space(3))) const void*)p;
}
// GLOBAL_LOAD_ASYNC_TO_LDS_B32: LDS[vdst_lane] = MEM[vaddr_lane] (ASYNCcnt).
__device__ __forceinline__ void async_ld_f32(unsigned ldsoff, const float* g) {
  asm volatile("global_load_async_to_lds_b32 %0, %1, off"
               :: "v"(ldsoff), "v"((unsigned long long)(uintptr_t)g)
               : "memory");
}
__device__ __forceinline__ void wait_async0() {
  asm volatile("s_wait_asynccnt 0" ::: "memory");
}

// ---------------------------------------------------------------------------
// Generic WMMA implicit-GEMM.
// MODE 0: conv1x3.  A = W[Cout][Cin*3], B = im2col(X) (pad 1 on s).
// MODE 1: conv_transpose_1x2. A rows m = k2*Cout+o from W[Cin][Cout][2];
//         store to (o, 2*s+k2).
// Tensors are (b, c, node, s); coff/noff select channel-/node-slices.
// S and all tile origins are multiples of 16 => a 16-column tile never
// crosses a (b, node) boundary, so column decomposition is done once.
// ---------------------------------------------------------------------------
struct GemmGeom {
  int M, K, Nsp, S, Sout;
  int in_ctot, in_coff, in_nnodes, in_noff;
  int out_ctot, out_coff, out_nnodes, out_noff;
  int Cout;
};

template <int MODE>
__global__ __launch_bounds__(256) void conv_gemm_wmma(
    const float* __restrict__ X, const float* __restrict__ W,
    const float* __restrict__ Bb, float* __restrict__ Y, GemmGeom g)
{
  __shared__ __align__(16) float    Asf[16][32];    // async landing zone (f32 weights)
  __shared__ __align__(32) _Float16 Af[32][16];     // A fragments, lane-major
  __shared__ __align__(32) _Float16 Bf[8][32][16];  // B fragments, lane-major / wave

  const int tid  = threadIdx.x;
  const int wv   = tid >> 5;
  const int lane = tid & 31;
  const int hi   = lane >> 4;            // lane half (wave32)
  const int l16  = lane & 15;
  const int m0   = blockIdx.y * 16;
  const int n0   = (blockIdx.x * 8 + wv) * 16;
  const int HSP  = NHALF * g.S;          // per-batch spatial extent (one half)

  // tile-column geometry (shared by all 16 columns of this wave's tile)
  const bool nok = (n0 < g.Nsp);
  int bq = 0, node = 0, sb = 0;
  if (nok) {
    bq = n0 / HSP;
    int rem = n0 - bq * HSP;
    node = rem / g.S;
    sb = rem - node * g.S;
  }
  const size_t cstride = (size_t)g.in_nnodes * g.S;
  const size_t colbase =
      (((size_t)bq * g.in_ctot + g.in_coff) * g.in_nnodes + g.in_noff + node) * g.S;

  v8f acc = {};

  for (int kk = 0; kk < g.K; kk += 32) {
    // ---- A tile: async DMA f32 weights into LDS (zeros for K padding) -----
    for (int e = tid; e < 512; e += 256) {
      int row = e >> 5, kl = e & 31;
      int kg = kk + kl;
      if (kg < g.K) {
        const float* src;
        int m = m0 + row;
        if (MODE == 0) {
          src = W + (size_t)m * g.K + kg;                    // (Cout, Cin*3)
        } else {
          int k2 = m / g.Cout, o = m - k2 * g.Cout;
          src = W + ((size_t)kg * g.Cout + o) * 2 + k2;      // (Cin, Cout, 2)
        }
        async_ld_f32(lds_off_u32(&Asf[row][kl]), src);
      } else {
        Asf[row][kl] = 0.f;
      }
    }
    // ---- B tile: batched gather -> f16 fragment-major ---------------------
    {
      int kl = lane, kg = kk + kl;
      int ci = (MODE == 0) ? (kg / 3) : kg;
      int dk = (MODE == 0) ? (kg - 3 * (kg / 3) - 1) : 0;
      const bool kok = (kg < g.K) && nok;
      const float* bp = X + colbase + (size_t)ci * cstride;
      float tv[16];
#pragma unroll
      for (int j = 0; j < 16; ++j) {
        int s2 = sb + j + dk;
        tv[j] = (kok && s2 >= 0 && s2 < g.S) ? bp[s2] : 0.f;
      }
      int jb = (kl >> 4) << 4, ii = kl & 15;
#pragma unroll
      for (int j = 0; j < 16; ++j) Bf[wv][jb + j][ii] = (_Float16)tv[j];
      // L2 prefetch for the next K-step's row of this column block
      if (nok && (kg + 32) < g.K) {
        int ci2 = (MODE == 0) ? ((kg + 32) / 3) : (kg + 32);
        __builtin_prefetch(X + colbase + (size_t)ci2 * cstride + sb, 0, 1);
      }
    }
    wait_async0();          // our async LDS writes landed
    __syncthreads();        // everyone's A-raw + B-frag tiles visible
    // ---- convert raw A -> fragment-major f16 ------------------------------
    for (int e = tid; e < 512; e += 256) {
      int lp = e >> 4, i = e & 15;
      int row = lp & 15, h2 = lp >> 4;
      int kl = 16 * h2 + i;
      Af[lp][i] = (_Float16)Asf[row][kl];
    }
    __syncthreads();
    // ---- fragments load as 2x ds_load_b128 each ---------------------------
    v16h af = *reinterpret_cast<const v16h*>(&Af[lane][0]);
    v16h bf = *reinterpret_cast<const v16h*>(&Bf[wv][lane][0]);
    acc = __builtin_amdgcn_wmma_f32_16x16x32_f16(false, af, false, bf,
                                                 (short)0, acc, false, false);
    __syncthreads();        // done reading LDS before next iteration restages
  }

  // ---- store D (8 rows per lane) + bias -----------------------------------
  if (nok) {
    const size_t ostride = (size_t)g.out_nnodes * g.Sout;
    const size_t obase =
        (((size_t)bq * g.out_ctot + g.out_coff) * g.out_nnodes + g.out_noff + node) *
        g.Sout;
    int s = sb + l16;
#pragma unroll
    for (int r = 0; r < 8; ++r) {
      int m = m0 + r + 8 * hi;
      float v = acc[r];
      if (MODE == 0) {
        v += Bb[m];
        Y[obase + (size_t)m * ostride + s] = v;
      } else {
        int k2 = m / g.Cout, o = m - k2 * g.Cout;
        v += Bb[o];
        Y[obase + (size_t)o * ostride + 2 * s + k2] = v;
      }
    }
  }
}

// ---------------------------------------------------------------------------
// Small fp32 kernels
// ---------------------------------------------------------------------------
__global__ void stem_conv_kernel(const float* __restrict__ x,
                                 const float* __restrict__ w,
                                 const float* __restrict__ bias,
                                 float* __restrict__ y, int total)
{
  int idx = blockIdx.x * 256 + threadIdx.x;
  if (idx >= total) return;
  int s = idx % S0_; int t = idx / S0_;
  int n = t % NN; t /= NN;
  int co = t % 2; int b = t / 2;
  float a = bias[co];
  for (int ci = 0; ci < 2; ++ci)
    for (int dk = 0; dk < 3; ++dk) {
      int s2 = s + dk - 1;
      if (s2 < 0 || s2 >= S0_) continue;
      a += x[(((size_t)b * 2 + ci) * NN + n) * S0_ + s2] * w[(co * 2 + ci) * 3 + dk];
    }
  y[(((size_t)b * 2 + co) * NN + n) * S0_ + s] = a;
}

struct GcnP {
  const float *b1, *b2, *b3, *bn_b, *bn_g, *ln1_b, *ln1_g, *ln2_b, *ln2_g,
              *scale, *w1, *w2, *w3;
};

__global__ void att_mlp_kernel(const float* __restrict__ ea, GcnP p,
                               float* __restrict__ att, int ne)
{
  int e = blockIdx.x * 256 + threadIdx.x;
  if (e >= ne) return;
  float h0[8];
#pragma unroll
  for (int i = 0; i < 8; ++i)
    h0[i] = ea[(size_t)e * 8 + i] * (p.bn_g[i] * BN_SC) + p.bn_b[i];

  float h1[16]; float m = 0.f;
  for (int j = 0; j < 16; ++j) {
    float a = p.b1[j];
    for (int i = 0; i < 8; ++i) a += h0[i] * p.w1[i * 16 + j];
    h1[j] = a; m += a;
  }
  m *= (1.f / 16.f);
  float v = 0.f;
  for (int j = 0; j < 16; ++j) { float d = h1[j] - m; v += d * d; }
  float rs = rsqrtf(v * (1.f / 16.f) + 1e-5f);
  for (int j = 0; j < 16; ++j)
    h1[j] = fmaxf((h1[j] - m) * rs * p.ln1_g[j] + p.ln1_b[j], 0.f);

  float h2[32]; m = 0.f;
  for (int j = 0; j < 32; ++j) {
    float a = p.b2[j];
    for (int i = 0; i < 16; ++i) a += h1[i] * p.w2[i * 32 + j];
    h2[j] = a; m += a;
  }
  m *= (1.f / 32.f);
  v = 0.f;
  for (int j = 0; j < 32; ++j) { float d = h2[j] - m; v += d * d; }
  rs = rsqrtf(v * (1.f / 32.f) + 1e-5f);
  for (int j = 0; j < 32; ++j)
    h2[j] = fmaxf((h2[j] - m) * rs * p.ln2_g[j] + p.ln2_b[j], 0.f);

  float o = p.b3[0];
  for (int j = 0; j < 32; ++j) o += h2[j] * p.w3[j];
  o = o > 0.f ? o : 0.2f * o;
  att[e] = 1.f / (1.f + expf(-o));
}

__global__ void rownorm_kernel(const float* __restrict__ X, float* __restrict__ out,
                               int ctot, int S, int D)
{
  int row = blockIdx.x;                       // b*NN + node
  int b = row / NN, node = row - b * NN;
  __shared__ float red[256];
  float s = 0.f;
  for (int d = threadIdx.x; d < D; d += 256) {
    int c = d / S, ss = d - c * S;
    float v = X[(((size_t)b * ctot + c) * NN + node) * S + ss];
    s += v * v;
  }
  red[threadIdx.x] = s; __syncthreads();
  for (int o = 128; o > 0; o >>= 1) {
    if (threadIdx.x < o) red[threadIdx.x] += red[threadIdx.x + o];
    __syncthreads();
  }
  if (threadIdx.x == 0) out[row] = sqrtf(red[0]);
}

// One block per dst node (edges come in fixed groups of DEG_ per dst) ->
// deterministic aggregation with no float atomics.
__global__ void gat_agg_kernel(const float* __restrict__ X,
                               const int* __restrict__ ei,
                               const float* __restrict__ att,
                               const float* __restrict__ hnorm,
                               const float* __restrict__ scale_p,
                               float* __restrict__ AGG, float* __restrict__ Yo,
                               int ctot, int S, int D, int etot)
{
  int row = blockIdx.x;                    // 0..B*NN-1
  __shared__ int   ssrc[DEG_];
  __shared__ float satt[DEG_];
  __shared__ float red[256];
  int e0 = row * DEG_;
  if (threadIdx.x < DEG_) {
    ssrc[threadIdx.x] = ei[e0 + threadIdx.x];
    satt[threadIdx.x] = att[e0 + threadIdx.x];
  }
  __syncthreads();
  int dstrow = ei[etot + e0];
  int b = dstrow / NN, node = dstrow - b * NN;

  float ssq = 0.f;
  for (int d = threadIdx.x; d < D; d += 256) {
    int c = d / S, s = d - c * S;
    float a = 0.f;
#pragma unroll
    for (int j = 0; j < DEG_; ++j) {
      int sr = ssrc[j]; int sb = sr / NN, sn = sr - sb * NN;
      a += satt[j] * X[(((size_t)sb * ctot + c) * NN + sn) * S + s];
    }
    AGG[(size_t)row * D + d] = a;
    ssq += a * a;
  }
  red[threadIdx.x] = ssq; __syncthreads();
  for (int o = 128; o > 0; o >>= 1) {
    if (threadIdx.x < o) red[threadIdx.x] += red[threadIdx.x + o];
    __syncthreads();
  }
  float sc = scale_p[0] * hnorm[dstrow] / fmaxf(sqrtf(red[0]), 1e-12f);
  for (int d = threadIdx.x; d < D; d += 256) {
    int c = d / S, s = d - c * S;
    Yo[(((size_t)b * ctot + c) * NN + node) * S + s] = sc * AGG[(size_t)row * D + d];
  }
}

__global__ void inorm_stats_kernel(const float* __restrict__ X,
                                   float* __restrict__ mean, float* __restrict__ rstd,
                                   int C, int ctot, int coff, int nnodes, int noff,
                                   int NHc, int S)
{
  int bc = blockIdx.x; int b = bc / C; int c = bc - b * C;
  const float* base = X + (((size_t)b * ctot + coff + c) * nnodes + noff) * S;
  int n = NHc * S;
  float s1 = 0.f, s2 = 0.f;
  for (int i = threadIdx.x; i < n; i += 256) { float v = base[i]; s1 += v; s2 += v * v; }
  __shared__ float r1[256], r2[256];
  r1[threadIdx.x] = s1; r2[threadIdx.x] = s2; __syncthreads();
  for (int o = 128; o > 0; o >>= 1) {
    if (threadIdx.x < o) { r1[threadIdx.x] += r1[threadIdx.x + o]; r2[threadIdx.x] += r2[threadIdx.x + o]; }
    __syncthreads();
  }
  if (threadIdx.x == 0) {
    float m = r1[0] / n;
    mean[bc] = m;
    rstd[bc] = rsqrtf(r2[0] / n - m * m + 1e-5f);
  }
}

__global__ void inorm_apply_kernel(const float* __restrict__ X,
                                   const float* __restrict__ res,
                                   float* __restrict__ Y,
                                   const float* __restrict__ mean,
                                   const float* __restrict__ rstd,
                                   int C, int in_ctot, int in_coff, int in_nnodes, int in_noff,
                                   int out_ctot, int out_coff, int out_nnodes, int out_noff,
                                   int res_ctot, int res_coff,
                                   int NHc, int S, int relu, int total)
{
  int idx = blockIdx.x * 256 + threadIdx.x;
  if (idx >= total) return;
  int s = idx % S; int t = idx / S;
  int node = t % NHc; t /= NHc;
  int c = t % C; int b = t / C;
  float v = (X[(((size_t)b * in_ctot + in_coff + c) * in_nnodes + in_noff + node) * S + s]
             - mean[b * C + c]) * rstd[b * C + c];
  if (res)
    v += res[(((size_t)b * res_ctot + res_coff + c) * in_nnodes + in_noff + node) * S + s];
  if (relu) v = fmaxf(v, 0.f);
  Y[(((size_t)b * out_ctot + out_coff + c) * out_nnodes + out_noff + node) * S + s] = v;
}

__global__ void stem_concat_kernel(const float* __restrict__ x,
                                   const float* __restrict__ x0g,
                                   const float* __restrict__ mean,
                                   const float* __restrict__ rstd,
                                   const float* __restrict__ g,
                                   const float* __restrict__ bb,
                                   float* __restrict__ xs0, int total)
{
  int idx = blockIdx.x * 256 + threadIdx.x;
  if (idx >= total) return;
  int s = idx % S0_; int t = idx / S0_;
  int n = t % NN; t /= NN;
  int c = t % 2; int b = t / 2;
  size_t src = (((size_t)b * 2 + c) * NN + n) * S0_ + s;
  xs0[(((size_t)b * 4 + c) * NN + n) * S0_ + s] = x[src];
  float v = (x0g[src] - mean[b]) * rstd[b] * g[((size_t)c * NN + n) * S0_ + s]
            + bb[((size_t)c * NN + n) * S0_ + s];
  xs0[(((size_t)b * 4 + 2 + c) * NN + n) * S0_ + s] = fmaxf(v, 0.f);
}

__global__ void elupool_kernel(const float* __restrict__ hs, float* __restrict__ Y,
                               int C3, int Si, int oct, int noff, int total)
{
  int idx = blockIdx.x * 256 + threadIdx.x;
  if (idx >= total) return;
  int So = Si >> 1;
  int so = idx % So; int t = idx / So;
  int node = t % NHALF; t /= NHALF;
  int c = t % C3; int b = t / C3;
  size_t ib = (((size_t)b * C3 + c) * NHALF + node) * Si + 2 * so;
  float a = eluf(hs[ib]), d = eluf(hs[ib + 1]);
  Y[(((size_t)b * oct + c) * NN + noff + node) * So + so] = fmaxf(a, d);
}

__global__ void copych_kernel(const float* __restrict__ X, float* __restrict__ Y,
                              int C, int ict, int icf, int oct, int ocf,
                              int nnodes, int S, int total)
{
  int idx = blockIdx.x * 256 + threadIdx.x;
  if (idx >= total) return;
  int s = idx % S; int t = idx / S;
  int node = t % nnodes; t /= nnodes;
  int c = t % C; int b = t / C;
  Y[(((size_t)b * oct + ocf + c) * nnodes + node) * S + s] =
      X[(((size_t)b * ict + icf + c) * nnodes + node) * S + s];
}

__global__ void final_kernel(const float* __restrict__ X,
                             const float* __restrict__ fw,
                             const float* __restrict__ fb,
                             float* __restrict__ out, int total)
{
  int idx = blockIdx.x * 256 + threadIdx.x;
  if (idx >= total) return;
  int s = idx % S0_; int t = idx / S0_;
  int nl = t % NHALF; int b = t / NHALF;
  float a = fb[0];
  for (int c = 0; c < 68; ++c)
    a += X[(((size_t)b * 68 + c) * NN + NHALF + nl) * S0_ + s] * fw[c];
  out[idx] = a;
}

// ---------------------------------------------------------------------------
// Host orchestration
// ---------------------------------------------------------------------------
extern "C" void kernel_launch(void* const* d_in, const int* in_sizes, int n_in,
                              void* d_out, int out_size, void* d_ws, size_t ws_size,
                              hipStream_t stream)
{
  (void)in_sizes; (void)n_in; (void)out_size; (void)ws_size;

  // ---- unpack inputs (dict order; params pytree: dicts sorted, lists ordered)
  int ix = 0;
  auto F = [&](int i) { return (const float*)d_in[i]; };
  const float* x_in      = F(ix++);
  const float* edge_attr = F(ix++);
  const float *dl_b[3][3], *dl_w[3][3], *dr_b[3][3], *dr_w[3][3];
  for (int l = 0; l < 3; ++l) for (int j = 0; j < 3; ++j) { dl_b[l][j] = F(ix++); dl_w[l][j] = F(ix++); }
  for (int l = 0; l < 3; ++l) for (int j = 0; j < 3; ++j) { dr_b[l][j] = F(ix++); dr_w[l][j] = F(ix++); }
  const float* final_b = F(ix++);
  const float* final_w = F(ix++);
  GcnP gdown[3], gstem, gup[3];
  auto RG = [&](GcnP& g) {
    g.b1 = F(ix++); g.b2 = F(ix++); g.b3 = F(ix++); g.bn_b = F(ix++); g.bn_g = F(ix++);
    g.ln1_b = F(ix++); g.ln1_g = F(ix++); g.ln2_b = F(ix++); g.ln2_g = F(ix++);
    g.scale = F(ix++); g.w1 = F(ix++); g.w2 = F(ix++); g.w3 = F(ix++);
  };
  for (int i = 0; i < 3; ++i) RG(gdown[i]);
  RG(gstem);
  for (int i = 0; i < 3; ++i) RG(gup[i]);
  const float* stem_b    = F(ix++);
  const float* stem_ln_b = F(ix++);
  const float* stem_ln_g = F(ix++);
  const float* stem_w    = F(ix++);
  const float *ul_b[3], *ul_w[3], *ur_b[3], *ur_w[3];
  for (int i = 0; i < 3; ++i) { ul_b[i] = F(ix++); ul_w[i] = F(ix++); }
  for (int i = 0; i < 3; ++i) { ur_b[i] = F(ix++); ur_w[i] = F(ix++); }
  const int* edge_index = (const int*)d_in[ix++];
  ix++; // batch (unused)

  // ---- workspace layout (fp32) ----
  float* wsp = (float*)d_ws;
  size_t off = 0;
  auto AL = [&](size_t n) { float* p = wsp + off; off += n; return p; };
  float* X0   = AL(153600);
  float* X0G  = AL(153600);
  float* XS0  = AL(307200);
  float* XS1  = AL(7372800);
  float* XS2  = AL(7372800);
  float* XS3  = AL(7372800);
  float* HSL  = AL(7372800);
  float* HSR  = AL(7372800);
  float* YB   = AL(7372800);
  float* YG   = AL(7372800);
  float* AGG  = AL(7372800);
  float* ATT  = AL(16384);
  float* HN   = AL(1024);
  float* MEAN = AL(2048);
  float* RSTD = AL(2048);
  float* UP0  = AL(12288000);
  float* UP1  = AL(12288000);
  float* UP2  = AL(5222400);
  float* TC   = AL(4915200);
  float* TCG  = AL(4915200);

  // ---- launch helpers ----
  auto gemm0 = [&](const float* X, const float* W, const float* Bb, float* Y,
                   int Cout, int Cin, int S,
                   int ict, int icf, int inn, int inoff,
                   int oct, int ocf, int onn, int onoff) {
    GemmGeom g;
    g.M = Cout; g.K = Cin * 3; g.Nsp = B_ * NHALF * S; g.S = S; g.Sout = S;
    g.in_ctot = ict; g.in_coff = icf; g.in_nnodes = inn; g.in_noff = inoff;
    g.out_ctot = oct; g.out_coff = ocf; g.out_nnodes = onn; g.out_noff = onoff;
    g.Cout = Cout;
    int tn = (g.Nsp + 15) / 16;
    dim3 grid((tn + 7) / 8, Cout / 16);
    conv_gemm_wmma<0><<<grid, 256, 0, stream>>>(X, W, Bb, Y, g);
  };
  auto gemm1 = [&](const float* X, const float* W, const float* Bb, float* Y,
                   int Cout, int Cin, int S,
                   int ict, int icf, int inn, int inoff,
                   int oct, int ocf, int onn, int onoff) {
    GemmGeom g;
    g.M = 2 * Cout; g.K = Cin; g.Nsp = B_ * NHALF * S; g.S = S; g.Sout = 2 * S;
    g.in_ctot = ict; g.in_coff = icf; g.in_nnodes = inn; g.in_noff = inoff;
    g.out_ctot = oct; g.out_coff = ocf; g.out_nnodes = onn; g.out_noff = onoff;
    g.Cout = Cout;
    int tn = (g.Nsp + 15) / 16;
    dim3 grid((tn + 7) / 8, (2 * Cout) / 16);
    conv_gemm_wmma<1><<<grid, 256, 0, stream>>>(X, W, Bb, Y, g);
  };
  auto istats = [&](const float* X, int C, int ctot, int coff, int nnodes, int noff,
                    int NHc, int S) {
    inorm_stats_kernel<<<B_ * C, 256, 0, stream>>>(X, MEAN, RSTD, C, ctot, coff,
                                                   nnodes, noff, NHc, S);
  };
  auto iapply = [&](const float* X, const float* res, float* Y, int C,
                    int ict, int icf, int inn, int inoff,
                    int oct, int ocf, int onn, int onoff,
                    int rct, int rcf, int NHc, int S, int relu) {
    int total = B_ * C * NHc * S;
    inorm_apply_kernel<<<(total + 255) / 256, 256, 0, stream>>>(
        X, res, Y, MEAN, RSTD, C, ict, icf, inn, inoff, oct, ocf, onn, onoff,
        rct, rcf, NHc, S, relu, total);
  };
  auto gnn = [&](const GcnP& gp, const float* Xb, float* Yb, int ctot, int S) {
    int D = ctot * S;
    att_mlp_kernel<<<(E_TOT + 255) / 256, 256, 0, stream>>>(edge_attr, gp, ATT, E_TOT);
    rownorm_kernel<<<B_ * NN, 256, 0, stream>>>(Xb, HN, ctot, S, D);
    gat_agg_kernel<<<B_ * NN, 256, 0, stream>>>(Xb, edge_index, ATT, HN, gp.scale,
                                                AGG, Yb, ctot, S, D, E_TOT);
  };

  // ======================= stem =======================
  {
    int tot0 = B_ * 2 * NN * S0_;
    stem_conv_kernel<<<(tot0 + 255) / 256, 256, 0, stream>>>(x_in, stem_w, stem_b, X0, tot0);
    gnn(gstem, X0, X0G, 2, S0_);
    // per-batch mean/var over (c,n,s): contiguous 600*128 per batch
    inorm_stats_kernel<<<B_, 256, 0, stream>>>(X0G, MEAN, RSTD, 1, 1, 0, 600, 0, 600, S0_);
    stem_concat_kernel<<<(tot0 + 255) / 256, 256, 0, stream>>>(
        x_in, X0G, MEAN, RSTD, stem_ln_g, stem_ln_b, XS0, tot0);
  }

  // ======================= down path =======================
  {
    const float* xin[3]  = {XS0, XS1, XS2};
    int   xin_ct[3]      = {4, 192, 384};
    float* xout[3]       = {XS1, XS2, XS3};
    int   Cin_[3]        = {4, 192, 384};
    int   Cc_[3]         = {64, 128, 256};
    int   Sin_[3]        = {128, 64, 32};
    for (int k = 0; k < 3; ++k) {
      int Cc = Cc_[k], Si = Sin_[k], So = Si / 2, C3 = 3 * Cc;
      for (int side = 0; side < 2; ++side) {
        float* hs = side ? HSR : HSL;
        int noff = side ? NHALF : 0;
        for (int j = 0; j < 3; ++j) {
          const float* Wj = side ? dr_w[k][j] : dl_w[k][j];
          const float* Bj = side ? dr_b[k][j] : dl_b[k][j];
          int cin = (j == 0) ? Cin_[k] : Cc;
          const float* Xs = (j == 0) ? xin[k] : hs;
          int ict  = (j == 0) ? xin_ct[k] : C3;
          int icf  = (j == 0) ? 0 : (j - 1) * Cc;
          int inn  = (j == 0) ? NN : NHALF;
          int inoff = (j == 0) ? noff : 0;
          gemm0(Xs, Wj, Bj, hs, Cc, cin, Si, ict, icf, inn, inoff, C3, j * Cc, NHALF, 0);
          istats(hs, Cc, C3, j * Cc, NHALF, 0, NHALF, Si);
          const float* res = (j == 0) ? nullptr : hs;
          iapply(hs, res, hs, Cc, C3, j * Cc, NHALF, 0, C3, j * Cc, NHALF, 0,
                 C3, (j == 0 ? 0 : (j - 1) * Cc), NHALF, Si, 0);
        }
        int totp = B_ * C3 * NHALF * So;
        elupool_kernel<<<(totp + 255) / 256, 256, 0, stream>>>(hs, YB, C3, Si, C3, noff, totp);
      }
      gnn(gdown[k], YB, YG, C3, So);
      istats(YG, C3, C3, 0, NN, 0, NN, So);
      iapply(YG, nullptr, xout[k], C3, C3, 0, NN, 0, C3, 0, NN, 0, 0, 0, NN, So, 1);
    }
  }

  // ======================= up path =======================
  {
    const float* cur = XS3; int ctin = 768;
    float* nexts[3]      = {UP0, UP1, UP2};
    int    Cout_[3]      = {256, 128, 64};
    int    Sin_[3]       = {16, 32, 64};
    const float* skip[3] = {XS2, XS1, XS0};
    int    skipC[3]      = {384, 192, 4};
    for (int k = 0; k < 3; ++k) {
      int Co = Cout_[k], Si = Sin_[k], So = 2 * Si;
      int nct = Co + skipC[k];
      for (int side = 0; side < 2; ++side) {
        int noff = side ? NHALF : 0;
        const float* Wk = side ? ur_w[k] : ul_w[k];
        const float* Bk = side ? ur_b[k] : ul_b[k];
        gemm1(cur, Wk, Bk, TC, Co, ctin, Si, ctin, 0, NN, noff, Co, 0, NN, noff);
      }
      gnn(gup[k], TC, TCG, Co, So);
      istats(TCG, Co, Co, 0, NN, 0, NN, So);
      iapply(TCG, nullptr, nexts[k], Co, Co, 0, NN, 0, nct, 0, NN, 0, 0, 0, NN, So, 1);
      int totc = B_ * skipC[k] * NN * So;
      copych_kernel<<<(totc + 255) / 256, 256, 0, stream>>>(
          skip[k], nexts[k], skipC[k], skipC[k], 0, nct, Co, NN, So, totc);
      cur = nexts[k]; ctin = nct;
    }
  }

  // ======================= final projection =======================
  {
    int tot = B_ * NHALF * S0_;
    final_kernel<<<(tot + 255) / 256, 256, 0, stream>>>(UP2, final_w, final_b,
                                                        (float*)d_out, tot);
  }
}
